// ReductionAttention_51187420234287
// MI455X (gfx1250) — matrix operand order repaired and checked by reading, not compiled, not run on hardware
//
#include <hip/hip_runtime.h>
#include <hip/hip_bf16.h>
#include <stdint.h>

// ---------------------------------------------------------------------------
// ReductionAttention for MI455X (gfx1250, wave32)
//   B=4, N=2048, DIM=512, HEADS=8, HEAD_DIM=64, M=4 query rows, SCALE=0.125
//   Dominant work: kv = x @ Wkv  (8192x512 @ 512x1024, 8.6 GFLOP)
//   -> bf16 WMMA (v_wmma_f32_16x16x32_bf16) + async global->LDS staging,
//      everything else scalar fp32 (1% of the FLOPs, L2-resident data).
// ---------------------------------------------------------------------------

typedef __attribute__((ext_vector_type(16))) __bf16 v16bf;
typedef __attribute__((ext_vector_type(8)))  float  v8f;
typedef unsigned short u16;
typedef int v4i_vs __attribute__((__vector_size__(16)));   // builtin's vector type

#define __AS1 __attribute__((address_space(1)))
#define __AS3 __attribute__((address_space(3)))

union Frag { uint4 q[2]; v16bf v; };

#ifndef __has_builtin
#define __has_builtin(x) 0
#endif

#if __has_builtin(__builtin_amdgcn_global_load_async_to_lds_b128)
#define ASYNC_LDS 1
#else
#define ASYNC_LDS 0
#endif

__device__ __forceinline__ u16 f32_to_bf16(float f) {
    union { float f; unsigned u; } v; v.f = f;
    unsigned u = v.u;
    u += 0x7fffu + ((u >> 16) & 1u);           // round-to-nearest-even
    return (u16)(u >> 16);
}
__device__ __forceinline__ float bf16_to_f32(u16 h) {
    union { unsigned u; float f; } v; v.u = ((unsigned)h) << 16;
    return v.f;
}
// unpack a packed pair of bf16 (as one u32) into two floats
__device__ __forceinline__ void bf16x2_to_f32(unsigned u, float& lo, float& hi) {
    union { unsigned u; float f; } a, b;
    a.u = u << 16;
    b.u = u & 0xffff0000u;
    lo = a.f; hi = b.f;
}

#if ASYNC_LDS
// GLOBAL_LOAD_ASYNC_TO_LDS_B128: hardware copies 16B/lane global->LDS,
// tracked by ASYNCcnt. Signature: (v4i as1* src, v4i as3* dst, Ii off, Ii cpol)
__device__ __forceinline__ void async_copy_b128(const u16* g, u16* l) {
    __builtin_amdgcn_global_load_async_to_lds_b128(
        (v4i_vs __AS1 *)g,
        (v4i_vs __AS3 *)l,
        0, 0);
}
#endif

// ---------------------------------------------------------------------------
// 1) x (fp32, 8192x512) -> xb (bf16), 4 elems / thread via float4
// ---------------------------------------------------------------------------
__global__ void __launch_bounds__(256) cvt_x_kernel(const float* __restrict__ x,
                                                    u16* __restrict__ xb, int n4) {
    int i = blockIdx.x * 256 + threadIdx.x;
    if (i < n4) {
        float4 f = ((const float4*)x)[i];
        uint2 o;
        o.x = (unsigned)f32_to_bf16(f.x) | ((unsigned)f32_to_bf16(f.y) << 16);
        o.y = (unsigned)f32_to_bf16(f.z) | ((unsigned)f32_to_bf16(f.w) << 16);
        ((uint2*)xb)[i] = o;
    }
}

// ---------------------------------------------------------------------------
// 2) Wkv (fp32, 512x1024 row-major (k,n)) -> wkvT (bf16, 1024x512 (n,k))
// ---------------------------------------------------------------------------
__global__ void __launch_bounds__(256) cvt_wkvT_kernel(const float* __restrict__ Wkv,
                                                       u16* __restrict__ wkvT) {
    int i = blockIdx.x * 256 + threadIdx.x;      // 0 .. 512*1024-1
    int k = i >> 10;
    int n = i & 1023;
    wkvT[(size_t)n * 512 + k] = f32_to_bf16(Wkv[i]);
}

// ---------------------------------------------------------------------------
// 3) kv GEMM: kvb[r, c] = sum_k xb[r,k] * WkvT[c,k], r<8192, c<1024, K=512
//    WG = 256 threads (8 waves as 2x4). Block tile 128x128; wave tile 64x32.
// ---------------------------------------------------------------------------
__global__ void __launch_bounds__(256) gemm_kv_kernel(const u16* __restrict__ xb,
                                                      const u16* __restrict__ wkvT,
                                                      u16* __restrict__ kvb) {
    __shared__ __align__(16) u16 As[128][40];   // 128 rows x 32 K (pad->40)
    __shared__ __align__(16) u16 Bs[128][40];   // 128 cols x 32 K (pad->40)

    const int t     = threadIdx.x;
    const int lane  = t & 31;
    const int wave  = t >> 5;
    const int waveM = wave >> 2;                // 0..1  (rows)
    const int waveN = wave & 3;                 // 0..3  (cols)
    const int rowBase = blockIdx.y * 128;
    const int colBase = blockIdx.x * 128;

    v8f acc[4][2];
    #pragma unroll
    for (int i = 0; i < 4; i++)
        #pragma unroll
        for (int j = 0; j < 2; j++) acc[i][j] = (v8f)(0.0f);

    // staging: each thread copies 16 bf16 (32B) of A and of B per K-step
    const int sRow = t >> 1;                    // 0..127
    const int sK   = (t & 1) * 16;              // 0 or 16 (elements)

    // fragment geometry (ISA 7.12.2, wave32)
    const int arow  = lane & 15;                // A: M row within tile
    const int ahalf = (lane >> 4) * 8;          // A: K half (0 / 8)
    const int bn    = lane & 15;                // B: N col within tile
    const int bk    = (lane >> 4) * 8;          // B: K half (0 / 8)

    for (int k0 = 0; k0 < 512; k0 += 32) {
        const u16* gA = xb   + (size_t)(rowBase + sRow) * 512 + k0 + sK;
        const u16* gB = wkvT + (size_t)(colBase + sRow) * 512 + k0 + sK;
#if ASYNC_LDS
        // hardware DMA into LDS, tracked by ASYNCcnt (no VGPR round-trip)
        async_copy_b128(gA,     &As[sRow][sK]);
        async_copy_b128(gA + 8, &As[sRow][sK + 8]);
        async_copy_b128(gB,     &Bs[sRow][sK]);
        async_copy_b128(gB + 8, &Bs[sRow][sK + 8]);
        asm volatile("s_wait_asynccnt 0" ::: "memory");
#else
        {
            const uint4* sa = (const uint4*)gA;
            const uint4* sb = (const uint4*)gB;
            uint4 a0 = sa[0], a1 = sa[1];
            uint4 b0 = sb[0], b1 = sb[1];
            *(uint4*)&As[sRow][sK]     = a0;
            *(uint4*)&As[sRow][sK + 8] = a1;
            *(uint4*)&Bs[sRow][sK]     = b0;
            *(uint4*)&Bs[sRow][sK + 8] = b1;
        }
#endif
        __syncthreads();

        Frag a[4];
        #pragma unroll
        for (int mt = 0; mt < 4; mt++) {
            const u16* ap = &As[waveM * 64 + mt * 16 + arow][0];
            a[mt].q[0] = *(const uint4*)(ap + ahalf);        // K = ahalf .. +7
            a[mt].q[1] = *(const uint4*)(ap + ahalf + 16);   // K = ahalf+16 .. +23
        }
        Frag b[2];
        #pragma unroll
        for (int nt = 0; nt < 2; nt++) {
            const u16* bp = &Bs[waveN * 32 + nt * 16 + bn][0];
            b[nt].q[0] = *(const uint4*)(bp + bk);           // K = bk .. bk+7
            b[nt].q[1] = *(const uint4*)(bp + bk + 16);      // K = bk+16 .. bk+23
        }
        #pragma unroll
        for (int mt = 0; mt < 4; mt++)
            #pragma unroll
            for (int nt = 0; nt < 2; nt++)
                acc[mt][nt] = __builtin_amdgcn_wmma_f32_16x16x32_bf16(
                    false, a[mt].v, false, b[nt].v, (short)0, acc[mt][nt],
                    false, false);
        __syncthreads();
    }

    // store D: lane&15 = N col; lane>>4 selects rows 0-7 / 8-15 across 8 VGPRs
    const int scol  = lane & 15;
    const int srow8 = (lane >> 4) * 8;
    #pragma unroll
    for (int mt = 0; mt < 4; mt++)
        #pragma unroll
        for (int nt = 0; nt < 2; nt++) {
            const int gr0 = rowBase + waveM * 64 + mt * 16 + srow8;
            const int gc  = colBase + waveN * 32 + nt * 16 + scol;
            #pragma unroll
            for (int r = 0; r < 8; r++)
                kvb[(size_t)(gr0 + r) * 1024 + gc] = f32_to_bf16(acc[mt][nt][r]);
        }
}

// ---------------------------------------------------------------------------
// 4) q = SCALE * (x[b, m] @ Wq)  -> qbuf (B*4 x 512 fp32).  16 blocks.
// ---------------------------------------------------------------------------
__global__ void __launch_bounds__(256) qproj_kernel(const float* __restrict__ x,
                                                    const float* __restrict__ Wq,
                                                    float* __restrict__ qbuf) {
    const int bm = blockIdx.x;                  // b*4 + m
    const int b = bm >> 2, m = bm & 3;
    __shared__ float sx[512];
    const int t = threadIdx.x;
    sx[t]       = x[(size_t)(b * 2048 + m) * 512 + t];
    sx[t + 256] = x[(size_t)(b * 2048 + m) * 512 + t + 256];
    __syncthreads();
    for (int c = t; c < 512; c += 256) {
        float acc = 0.0f;
        for (int k = 0; k < 512; k++) acc += sx[k] * Wq[(size_t)k * 512 + c];
        qbuf[(size_t)bm * 512 + c] = acc * 0.125f;   // SCALE = 64^-0.5
    }
}

// ---------------------------------------------------------------------------
// 5) attention per (b,h): dots = q k^T + bias[:, :, 0:4]; softmax; attn @ v
//    32 blocks x 256 threads. k/v read as bf16 from kvb (L2-resident).
// ---------------------------------------------------------------------------
__global__ void __launch_bounds__(256) attn_kernel(const float* __restrict__ qbuf,
                                                   const u16* __restrict__ kvb,
                                                   const float* __restrict__ bias,
                                                   float* __restrict__ ohbuf) {
    const int bh = blockIdx.x;
    const int b = bh >> 3, h = bh & 7;
    __shared__ float sdots[2048];
    __shared__ float sq[4][64];
    __shared__ float red[256];
    const int t = threadIdx.x;
    {
        int m = t >> 6, d = t & 63;
        sq[m][d] = qbuf[(size_t)(b * 4 + m) * 512 + h * 64 + d];
    }
    __syncthreads();

    for (int m = 0; m < 4; m++) {
        // dots + bias, track max
        float lmax = -1e30f;
        #pragma unroll
        for (int i = 0; i < 8; i++) {
            const int n = t + i * 256;
            // k row: 64 bf16 = 8x uint4 (128B aligned), vectorized loads
            const uint4* kr4 = (const uint4*)(kvb + (size_t)(b * 2048 + n) * 1024 + h * 64);
            float dot = 0.0f;
            #pragma unroll
            for (int c = 0; c < 8; c++) {
                uint4 p = kr4[c];
                float f0, f1;
                bf16x2_to_f32(p.x, f0, f1); dot += sq[m][c*8+0]*f0 + sq[m][c*8+1]*f1;
                bf16x2_to_f32(p.y, f0, f1); dot += sq[m][c*8+2]*f0 + sq[m][c*8+3]*f1;
                bf16x2_to_f32(p.z, f0, f1); dot += sq[m][c*8+4]*f0 + sq[m][c*8+5]*f1;
                bf16x2_to_f32(p.w, f0, f1); dot += sq[m][c*8+6]*f0 + sq[m][c*8+7]*f1;
            }
            dot += bias[(((size_t)(b * 8 + h)) * 2048 + m) * 2048 + n];
            sdots[n] = dot;
            lmax = fmaxf(lmax, dot);
        }
        red[t] = lmax; __syncthreads();
        for (int s = 128; s > 0; s >>= 1) {
            if (t < s) red[t] = fmaxf(red[t], red[t + s]);
            __syncthreads();
        }
        const float mx = red[0]; __syncthreads();

        float lsum = 0.0f;
        #pragma unroll
        for (int i = 0; i < 8; i++) {
            const int n = t + i * 256;
            float e = __expf(sdots[n] - mx);
            sdots[n] = e;
            lsum += e;
        }
        red[t] = lsum; __syncthreads();
        for (int s = 128; s > 0; s >>= 1) {
            if (t < s) red[t] += red[t + s];
            __syncthreads();
        }
        const float inv = 1.0f / red[0]; __syncthreads();

        // attn @ v : 4 groups of 64 threads; group g covers n in [g*512, g*512+512)
        const int g = t >> 6, d = t & 63;
        float part = 0.0f;
        const u16* vcol = kvb + (size_t)(b * 2048) * 1024 + 512 + h * 64 + d;
        for (int n = g * 512; n < g * 512 + 512; n++)
            part += sdots[n] * bf16_to_f32(vcol[(size_t)n * 1024]);
        red[t] = part; __syncthreads();
        if (t < 64) {
            float o = (red[t] + red[t + 64] + red[t + 128] + red[t + 192]) * inv;
            ohbuf[(size_t)(b * 4 + m) * 512 + h * 64 + t] = o;
        }
        __syncthreads();
    }
}

// ---------------------------------------------------------------------------
// 6) out = ohbuf @ Wo + bo  -> d_out (16 x 512 fp32).  16 blocks.
// ---------------------------------------------------------------------------
__global__ void __launch_bounds__(256) oproj_kernel(const float* __restrict__ ohbuf,
                                                    const float* __restrict__ Wo,
                                                    const float* __restrict__ bo,
                                                    float* __restrict__ out) {
    const int bm = blockIdx.x;
    __shared__ float so[512];
    const int t = threadIdx.x;
    so[t]       = ohbuf[(size_t)bm * 512 + t];
    so[t + 256] = ohbuf[(size_t)bm * 512 + t + 256];
    __syncthreads();
    for (int c = t; c < 512; c += 256) {
        float acc = bo[c];
        for (int k = 0; k < 512; k++) acc += so[k] * Wo[(size_t)k * 512 + c];
        out[(size_t)bm * 512 + c] = acc;
    }
}

// ---------------------------------------------------------------------------
extern "C" void kernel_launch(void* const* d_in, const int* in_sizes, int n_in,
                              void* d_out, int out_size, void* d_ws, size_t ws_size,
                              hipStream_t stream) {
    const float* x    = (const float*)d_in[0];   // (4, 2048, 512)
    const float* bias = (const float*)d_in[1];   // (4, 8, 2048, 2048)
    const float* Wq   = (const float*)d_in[2];   // (512, 512)
    const float* Wkv  = (const float*)d_in[3];   // (512, 1024)
    const float* Wo   = (const float*)d_in[4];   // (512, 512)
    const float* bo   = (const float*)d_in[5];   // (512,)
    float* out = (float*)d_out;                  // (4, 4, 512)

    // workspace layout
    char* ws = (char*)d_ws;
    u16*   xb    = (u16*)ws;                                   // 8192*512  bf16  (8.0 MiB)
    u16*   wkvT  = (u16*)(ws + (size_t)8192 * 512 * 2);        // 1024*512  bf16  (1.0 MiB)
    u16*   kvb   = (u16*)(ws + (size_t)8192 * 512 * 2
                             + (size_t)1024 * 512 * 2);        // 8192*1024 bf16 (16.0 MiB)
    float* qbuf  = (float*)(ws + (size_t)8192 * 512 * 2
                               + (size_t)1024 * 512 * 2
                               + (size_t)8192 * 1024 * 2);     // 16*512 f32
    float* ohbuf = qbuf + 16 * 512;                            // 16*512 f32

    // 1) x -> bf16 (1,048,576 float4s)
    cvt_x_kernel<<<4096, 256, 0, stream>>>(x, xb, 8192 * 512 / 4);
    // 2) Wkv -> bf16 transposed (524,288 elems)
    cvt_wkvT_kernel<<<2048, 256, 0, stream>>>(Wkv, wkvT);
    // 3) big GEMM via WMMA: grid = (1024/128 cols, 8192/128 rows)
    gemm_kv_kernel<<<dim3(8, 64), 256, 0, stream>>>(xb, wkvT, kvb);
    // 4) q projection (scaled)
    qproj_kernel<<<16, 256, 0, stream>>>(x, Wq, qbuf);
    // 5) attention (bias gather + softmax + attn@v)
    attn_kernel<<<32, 256, 0, stream>>>(qbuf, kvb, bias, ohbuf);
    // 6) output projection
    oproj_kernel<<<16, 256, 0, stream>>>(ohbuf, Wo, bo, out);
}
